// Perception_59304908423676
// MI455X (gfx1250) — matrix operand order repaired and checked by reading, not compiled, not run on hardware
//
#include <hip/hip_runtime.h>

typedef _Float16 v16h  __attribute__((ext_vector_type(16)));
typedef float    v8f   __attribute__((ext_vector_type(8)));
typedef unsigned u32x4 __attribute__((ext_vector_type(4)));
typedef float    f32x4 __attribute__((ext_vector_type(4)));

struct Frag16 { union { v16h v; u32x4 q[2]; _Float16 h[16]; }; };
struct FragAcc { union { v8f v; float f[8]; }; };

#define Bsz 4
#define Nn  4096
#define Dd  512
#define MTOT (Bsz * Nn)   // 16384

// ---------------------------------------------------------------------------
// d_inv[b,j] = rsqrt(sum_i adj[b,i,j])   (column sums, coalesced across j)
// ---------------------------------------------------------------------------
__global__ __launch_bounds__(256) void colsum_rsqrt_kernel(
    const float* __restrict__ adj, float* __restrict__ dinv)
{
    int g = blockIdx.x * 256 + threadIdx.x;          // 0 .. B*N-1
    int b = g >> 12;
    int j = g & (Nn - 1);
    const float* base = adj + (((size_t)b) << 24) + j;
    float s = 0.f;
    for (int i = 0; i < Nn; ++i) s += base[((size_t)i) << 12];
    dinv[g] = (s > 0.f) ? rsqrtf(fmaxf(s, 1e-30f)) : 0.f;
}

// fp32 -> fp16, straight copy
__global__ __launch_bounds__(256) void f32_to_f16_kernel(
    const float* __restrict__ src, _Float16* __restrict__ dst, int n)
{
    int g = blockIdx.x * 256 + threadIdx.x;
    if (g < n) dst[g] = (_Float16)src[g];
}

// fp32 -> fp16 with 512x512 transpose (for weights): dst[c][r] = src[r][c]
__global__ __launch_bounds__(256) void f32_to_f16_t512_kernel(
    const float* __restrict__ src, _Float16* __restrict__ dst)
{
    int g = blockIdx.x * 256 + threadIdx.x;
    int r = g >> 9, c = g & 511;
    dst[c * 512 + r] = (_Float16)src[g];
}

// ---------------------------------------------------------------------------
// WMMA GEMM. Block: 256 thr (8 waves), C tile 256x128, wave tile 64x64
// (16 x v_wmma_f32_16x16x32_f16 per wave per K-step), BK = 32.
//
//  ADJ=false (EPI=0): Zt = (D * (Act @ Wt^T...)) stored TRANSPOSED:
//       A = Act [16384][512] f16 row-major, B(LDS n-major) from Wt[n][k],
//       epilogue scales row i by dinv[i] and writes Zt[n][i] via LDS bounce.
//  ADJ=true  (EPI=1|2): C(b) = D_i * (adj(b)^T @ Zt^T(b)) + bias (relu if 1):
//       A[i][k] = adj[b][k][i] (raw 0/1 -> exact f16), B from Zt[n][b*4096+k].
// ---------------------------------------------------------------------------
template <bool ADJ, int EPI>
__global__ __launch_bounds__(256) void gcn_gemm_kernel(
    const _Float16* __restrict__ Amat,   // !ADJ: activations [MTOT][512]
    const float*    __restrict__ adj,    //  ADJ: [B][N][N] fp32
    const float*    __restrict__ dinv,   // [MTOT]
    const _Float16* __restrict__ Bmat,   // !ADJ: Wt[512][512]; ADJ: Zt[512][MTOT]
    const float*    __restrict__ bias,   // [512] (EPI>0)
    _Float16*       __restrict__ out16,  // EPI0: Zt ; EPI1: Act
    float*          __restrict__ out32)  // EPI2: final output
{
    constexpr int LDA  = 40;             // 32 + pad halfs (80B rows, 16B aligned)
    constexpr int LDBK = 40;
    // staging: As 256*40*2=20480B, Bs 128*40*2=10240B -> 30720B
    // EPI0 epilogue: 8 waves * 64*68*2 = 69632B (reuses same smem)
    __shared__ __align__(16) char smem[(EPI == 0) ? 69632 : 30720];
    _Float16* As = (_Float16*)smem;
    _Float16* Bs = (_Float16*)(smem + 20480);

    const int tid   = threadIdx.x;
    const int lane  = tid & 31;
    const int wave  = tid >> 5;
    const int lhalf = lane & 15;
    const int hi    = lane >> 4;                 // 0 | 1
    const int waveM = wave & 3;                  // 4 waves along M
    const int waveN = wave >> 2;                 // 2 waves along N
    const int tileN0 = blockIdx.x * 128;
    const int row0   = blockIdx.y * 256;
    const int b      = ADJ ? (int)blockIdx.z : 0;
    const int K      = ADJ ? Nn : Dd;

    FragAcc acc[4][4];
#pragma unroll
    for (int m = 0; m < 4; ++m)
#pragma unroll
        for (int n = 0; n < 4; ++n)
#pragma unroll
            for (int e = 0; e < 8; ++e) acc[m][n].f[e] = 0.f;

    for (int k0 = 0; k0 < K; k0 += 32) {
        __syncthreads();
        // ---- stage A tile (256 i x 32 k) -----------------------------------
        if constexpr (ADJ) {
            // As[i][k] = adj[b][k0+k][row0+i]  (row-read of adj, LDS scatter)
            const int k = tid >> 3, chunk = tid & 7;
            const float* srcrow = adj + (((size_t)b) << 24) +
                                  (((size_t)(k0 + k)) << 12) + row0 + chunk * 32;
#pragma unroll
            for (int r = 0; r < 8; ++r) {
                f32x4 v = ((const f32x4*)srcrow)[r];
#pragma unroll
                for (int c = 0; c < 4; ++c)
                    As[(chunk * 32 + r * 4 + c) * LDA + k] = (_Float16)v[c];
            }
            if (k0 + 32 < K)
                __builtin_prefetch((const void*)(srcrow + ((size_t)32 << 12)), 0, 1);
        } else {
            // row-major f16: 4 x b128 load, 4 x b128 LDS store
            const _Float16* src = Amat + (((size_t)(row0 + tid)) << 9) + k0;
#pragma unroll
            for (int r = 0; r < 4; ++r)
                *(u32x4*)(As + tid * LDA + r * 8) = ((const u32x4*)src)[r];
        }
        // ---- stage B tile (128 n x 32 k), K-contiguous source --------------
        {
            const int n = tid >> 1, seg = tid & 1;
            const size_t bstride = ADJ ? (size_t)MTOT : (size_t)Dd;
            const size_t col0    = ADJ ? (size_t)((b << 12) + k0) : (size_t)k0;
            u32x4 v = *(const u32x4*)(Bmat + (size_t)(tileN0 + n) * bstride +
                                      col0 + seg * 16);
            *(u32x4*)(Bs + n * LDBK + seg * 16) = v;
        }
        __syncthreads();

        // ---- fragments: 2 x ds_load_b128 each ------------------------------
        Frag16 af[4], bf[4];
#pragma unroll
        for (int m = 0; m < 4; ++m) {
            const _Float16* p = As + (waveM * 64 + m * 16 + lhalf) * LDA + hi * 8;
            af[m].q[0] = *(const u32x4*)p;
            af[m].q[1] = *(const u32x4*)(p + 16);
        }
#pragma unroll
        for (int n = 0; n < 4; ++n) {
            const _Float16* p = Bs + (waveN * 64 + n * 16 + lhalf) * LDBK + hi * 8;
            bf[n].q[0] = *(const u32x4*)p;
            bf[n].q[1] = *(const u32x4*)(p + 16);
        }
#pragma unroll
        for (int m = 0; m < 4; ++m)
#pragma unroll
            for (int n = 0; n < 4; ++n)
                acc[m][n].v = __builtin_amdgcn_wmma_f32_16x16x32_f16(
                    false, af[m].v, false, bf[n].v,
                    (short)0, acc[m][n].v, false, false);
    }

    // ---- epilogue ----------------------------------------------------------
    if constexpr (EPI == 0) {
        // scale rows by dinv, transpose 64x64 wave tile via wave-private LDS,
        // store Zt[n][row] coalesced (b32 per lane)
        __syncthreads();
        _Float16* T = (_Float16*)smem + wave * (64 * 68);
#pragma unroll
        for (int m = 0; m < 4; ++m) {
            float dv[8];
#pragma unroll
            for (int e = 0; e < 8; ++e)
                dv[e] = dinv[row0 + waveM * 64 + m * 16 + hi * 8 + e];
#pragma unroll
            for (int n = 0; n < 4; ++n)
#pragma unroll
                for (int e = 0; e < 8; ++e)
                    T[(n * 16 + lhalf) * 68 + m * 16 + hi * 8 + e] =
                        (_Float16)(acc[m][n].f[e] * dv[e]);
        }
#pragma unroll 4
        for (int r = 0; r < 64; ++r) {
            unsigned w = *(const unsigned*)(T + r * 68 + lane * 2);
            const size_t gn = tileN0 + waveN * 64 + r;
            const size_t gi = row0 + waveM * 64 + lane * 2;
            *(unsigned*)(out16 + gn * MTOT + gi) = w;
        }
    } else {
#pragma unroll
        for (int m = 0; m < 4; ++m) {
            float dv[8];
#pragma unroll
            for (int e = 0; e < 8; ++e)
                dv[e] = dinv[(b << 12) + row0 + waveM * 64 + m * 16 + hi * 8 + e];
#pragma unroll
            for (int n = 0; n < 4; ++n) {
                const int gn = tileN0 + waveN * 64 + n * 16 + lhalf;
                const float bv = bias[gn];
#pragma unroll
                for (int e = 0; e < 8; ++e) {
                    const int gi = row0 + waveM * 64 + m * 16 + hi * 8 + e;
                    float val = acc[m][n].f[e] * dv[e] + bv;
                    const size_t oidx =
                        (((size_t)((b << 12) + gi)) << 9) + gn;
                    if constexpr (EPI == 1)
                        out16[oidx] = (_Float16)(val > 0.f ? val : 0.f);
                    else
                        out32[oidx] = val;
                }
            }
        }
    }
}

// ---------------------------------------------------------------------------
extern "C" void kernel_launch(void* const* d_in, const int* in_sizes, int n_in,
                              void* d_out, int out_size, void* d_ws, size_t ws_size,
                              hipStream_t stream) {
    const float* x   = (const float*)d_in[0];
    const float* adj = (const float*)d_in[1];
    const float* W1  = (const float*)d_in[2];
    const float* b1  = (const float*)d_in[3];
    const float* W2  = (const float*)d_in[4];
    const float* b2  = (const float*)d_in[5];
    const float* W3  = (const float*)d_in[6];
    const float* b3  = (const float*)d_in[7];
    float* out = (float*)d_out;

    char* ws = (char*)d_ws;
    float*    dinv = (float*)ws;                                  // 64 KB
    _Float16* Act  = (_Float16*)(ws + (1 << 16));                 // 16 MB
    _Float16* Zt   = (_Float16*)(ws + (1 << 16) + (1 << 24));     // 16 MB (transposed)
    _Float16* Wt   = (_Float16*)(ws + (1 << 16) + 2 * (1 << 24)); // 1.5 MB (transposed)

    const int NACT = Bsz * Nn * Dd;   // 8388608
    const int NW   = Dd * Dd;         // 262144

    f32_to_f16_kernel<<<NACT / 256, 256, 0, stream>>>(x, Act, NACT);
    f32_to_f16_t512_kernel<<<NW / 256, 256, 0, stream>>>(W1, Wt);
    f32_to_f16_t512_kernel<<<NW / 256, 256, 0, stream>>>(W2, Wt + NW);
    f32_to_f16_t512_kernel<<<NW / 256, 256, 0, stream>>>(W3, Wt + 2 * NW);
    colsum_rsqrt_kernel<<<(Bsz * Nn) / 256, 256, 0, stream>>>(adj, dinv);

    dim3 gG1(Dd / 128, MTOT / 256, 1);   // (4, 64, 1)
    dim3 gG2(Dd / 128, Nn / 256, Bsz);   // (4, 16, 4)

    // layer 1: Zt = T(D*(X@W1)) ; H1 = relu(D*(A^T@Zt^T) + b1)
    gcn_gemm_kernel<false, 0><<<gG1, 256, 0, stream>>>(Act, nullptr, dinv, Wt,          nullptr, Zt,  nullptr);
    gcn_gemm_kernel<true,  1><<<gG2, 256, 0, stream>>>(nullptr, adj, dinv, Zt,          b1,      Act, nullptr);
    // layer 2
    gcn_gemm_kernel<false, 0><<<gG1, 256, 0, stream>>>(Act, nullptr, dinv, Wt + NW,     nullptr, Zt,  nullptr);
    gcn_gemm_kernel<true,  1><<<gG2, 256, 0, stream>>>(nullptr, adj, dinv, Zt,          b2,      Act, nullptr);
    // layer 3 (fp32 out, no relu)
    gcn_gemm_kernel<false, 0><<<gG1, 256, 0, stream>>>(Act, nullptr, dinv, Wt + 2 * NW, nullptr, Zt,  nullptr);
    gcn_gemm_kernel<true,  2><<<gG2, 256, 0, stream>>>(nullptr, adj, dinv, Zt,          b3,      nullptr, out);
}
